// So3kratesBlock_73469710566111
// MI455X (gfx1250) — matrix operand order repaired and checked by reading, not compiled, not run on hardware
//
#include <hip/hip_runtime.h>
#include <cstdint>
#include <cstddef>

// ---------------------------------------------------------------------------
// So3krates forward for MI455X (gfx1250, wave32).
// All dense GEMMs run on V_WMMA_F32_16X16X32_F16 (f16 in, f32 acc).
// Segment-sums use global f32 atomics into L2-resident buffers.
// ---------------------------------------------------------------------------

typedef __attribute__((ext_vector_type(16))) _Float16 v16h;
typedef __attribute__((ext_vector_type(8)))  float    v8f;

constexpr int kN   = 30000;
constexpr int kE   = 480000;
constexpr int kC   = 128;
constexpr int kR   = 32;
constexpr int kHID = 64;

struct Params {
  const float* edge_vectors; const float* distances; const float* cutoffs;
  const int* species; const int* senders; const int* receivers;
  const float* embed;
  const float *Wk, *Wkg, *Wo, *Wq, *Wqg, *Wv;
  const float *fb_rad_W1, *fb_rad_W2, *fb_rad_b1, *fb_rad_b2;
  const float *fb_sph_W1, *fb_sph_W2, *fb_sph_b1, *fb_sph_b2;
  const float *gb_rad_W1, *gb_rad_W2, *gb_rad_b1, *gb_rad_b2;
  const float *gb_sph_W1, *gb_sph_W2, *gb_sph_b1, *gb_sph_b2;
  const float *ib_W, *ib_b, *out_W1, *out_W2, *out_b1, *out_b2;
};

struct Wsp {
  _Float16* ef16;      // [E,32]  f16 RBF features (WMMA A operand)
  float*    sh;        // [E,8]   spherical harmonics
  float*    nf;        // [N,128] node features f32
  _Float16* nf16;      // [N,128] node features f16
  float*    qkv;       // [N,384]
  float*    qg;        // [N,2]
  float*    kg;        // [N,2]
  float*    agg;       // [N,128] attention segment-sum (atomics)
  _Float16* agg16;     // [N,128]
  float*    chi;       // [N,8]
  float*    chi_nxt;   // [N,8]
  _Float16* fbW1t[2];  // [64,32]   W^T f16
  _Float16* gbW1t[2];  // [64,32]
  _Float16* fbW2t[2];  // [128,128] concat(rad_W2;sph_W2)^T
  _Float16* Wqkvt[2];  // [384,128]
  _Float16* Wot[2];    // [128,128]
  _Float16* outW1t;    // [128,128]
  float* fb_b2sum[2];  // [128]
  float* gb_b2sum[2];  // [2]
  float* gbW2cat[2];   // [128,2] f32
};

__device__ __forceinline__ float silu(float x) { return x / (1.0f + __expf(-x)); }

__device__ __forceinline__ v8f wmma16(v16h a, v16h b, v8f c) {
  // D = A(16x32 f16) * B(32x16 f16) + C(16x16 f32)
  return __builtin_amdgcn_wmma_f32_16x16x32_f16(false, a, false, b, (short)0, c,
                                                false, false);
}

// A fragment from row-major f16 [16 x ld], K-chunk starting at k0.
// ISA 16-bit A layout: lanes 0-15 hold K{0..7,16..23}, lanes 16-31 K{8..15,24..31}.
__device__ __forceinline__ v16h load_a16(const _Float16* base, int ld, int k0) {
  int lane = threadIdx.x & 31;
  const _Float16* q = base + (size_t)(lane & 15) * ld + k0 + ((lane >> 4) ? 8 : 0);
  v16h f;
#pragma unroll
  for (int i = 0; i < 8; ++i) { f[i] = q[i]; f[8 + i] = q[16 + i]; }
  return f;
}

// B fragment from pre-transposed weights W^T stored [Nout x ldk] row-major:
// lane n = column, lanes 0-15 hold K{0..7,16..23}, lanes 16-31 K{8..15,24..31}.
__device__ __forceinline__ v16h load_bt16(const _Float16* wt, int ldk, int ntile, int k0) {
  int lane = threadIdx.x & 31;
  const _Float16* q = wt + (size_t)(ntile * 16 + (lane & 15)) * ldk + k0 + ((lane >> 4) ? 8 : 0);
  v16h f;
#pragma unroll
  for (int i = 0; i < 8; ++i) { f[i] = q[i]; f[8 + i] = q[16 + i]; }
  return f;
}

// ---------------------------------------------------------------------------
// Weight prep: f16 conversion + transpose into WMMA-friendly layouts.
// ---------------------------------------------------------------------------
__global__ __launch_bounds__(256) void k_prep_weights(Params P, Wsp W) {
  const long PER_L = 86402;
  long idx = (long)blockIdx.x * 256 + threadIdx.x;
  if (idx < 2 * PER_L) {
    int l = (int)(idx / PER_L);
    long i = idx % PER_L;
    if (i < 2048) {
      int n = (int)(i / 32), k = (int)(i % 32);
      W.fbW1t[l][n * 32 + k] = (_Float16)P.fb_rad_W1[l * 2048 + k * 64 + n];
    } else if (i < 4096) {
      i -= 2048; int n = (int)(i / 32), k = (int)(i % 32);
      W.gbW1t[l][n * 32 + k] = (_Float16)P.gb_rad_W1[l * 2048 + k * 64 + n];
    } else if (i < 20480) {
      i -= 4096; int n = (int)(i / 128), k = (int)(i % 128);
      float v = (k < 64) ? P.fb_rad_W2[l * 8192 + k * 128 + n]
                         : P.fb_sph_W2[l * 8192 + (k - 64) * 128 + n];
      W.fbW2t[l][n * 128 + k] = (_Float16)v;
    } else if (i < 69632) {
      i -= 20480; int n = (int)(i / 128), k = (int)(i % 128);
      float v = (n < 128) ? P.Wq[l * 16384 + k * 128 + n]
              : (n < 256) ? P.Wk[l * 16384 + k * 128 + (n - 128)]
                          : P.Wv[l * 16384 + k * 128 + (n - 256)];
      W.Wqkvt[l][(long)n * 128 + k] = (_Float16)v;
    } else if (i < 86016) {
      i -= 69632; int n = (int)(i / 128), k = (int)(i % 128);
      W.Wot[l][n * 128 + k] = (_Float16)P.Wo[l * 16384 + k * 128 + n];
    } else if (i < 86144) {
      i -= 86016;
      W.fb_b2sum[l][i] = P.fb_rad_b2[l * 128 + i] + P.fb_sph_b2[l * 128 + i];
    } else if (i < 86146) {
      i -= 86144;
      W.gb_b2sum[l][i] = P.gb_rad_b2[l * 2 + i] + P.gb_sph_b2[l * 2 + i];
    } else {
      i -= 86146; int k = (int)(i / 2), j = (int)(i % 2);
      W.gbW2cat[l][k * 2 + j] = (k < 64) ? P.gb_rad_W2[l * 128 + k * 2 + j]
                                         : P.gb_sph_W2[l * 128 + (k - 64) * 2 + j];
    }
  } else if (idx < 2 * PER_L + 16384) {
    long i = idx - 2 * PER_L;
    int n = (int)(i / 128), k = (int)(i % 128);
    W.outW1t[n * 128 + k] = (_Float16)P.out_W1[k * 128 + n];
  }
}

// Node embedding + chi zero-init.
__global__ __launch_bounds__(256) void k_prep_nodes(Params P, Wsp W) {
  long idx = (long)blockIdx.x * 256 + threadIdx.x;
  if (idx >= (long)kN * kC) return;
  int n = (int)(idx / kC), c = (int)(idx % kC);
  float v = P.embed[(long)P.species[n] * kC + c];
  W.nf[idx] = v; W.nf16[idx] = (_Float16)v;
  if (c < 8) W.chi[(long)n * 8 + c] = 0.0f;
}

// RBF (f16), spherical harmonics, chi seed (atomic segment-sum / 16).
__global__ __launch_bounds__(256) void k_prep_edges(Params P, Wsp W) {
  int e = blockIdx.x * 256 + threadIdx.x;
  if (e >= kE) return;
  float d = P.distances[e];
  float inv = 1.0f / d;
  float x = P.edge_vectors[3 * e] * inv;
  float y = P.edge_vectors[3 * e + 1] * inv;
  float z = P.edge_vectors[3 * e + 2] * inv;
  const float s3 = 1.73205081f, s5 = 2.23606798f, s15 = 3.87298335f;
  float sh8[8] = { s3 * x, s3 * y, s3 * z,
                   s15 * x * y, s15 * y * z, 0.5f * s5 * (3.f * z * z - 1.f),
                   s15 * x * z, 0.5f * s15 * (x * x - y * y) };
  float cut = P.cutoffs[e];
  int r = P.receivers[e];
#pragma unroll
  for (int c = 0; c < 8; ++c) {
    W.sh[(long)e * 8 + c] = sh8[c];
    atomicAdd(&W.chi[(long)r * 8 + c], sh8[c] * cut * (1.0f / 16.0f));
  }
  float t = __expf(-d);
  const float E5 = 0.00673794700f;
  const float STEP = (1.0f - E5) / 31.0f;
  const float G = (2.0f / 32.0f) * (1.0f - E5);
  const float BETA = 1.0f / (G * G);
#pragma unroll
  for (int i = 0; i < kR; ++i) {
    float dd = t - (E5 + i * STEP);
    W.ef16[(long)e * kR + i] = (_Float16)__expf(-BETA * dd * dd);
  }
}

__global__ __launch_bounds__(256) void k_begin_layer(Wsp W) {
  long idx = (long)blockIdx.x * 256 + threadIdx.x;
  if (idx >= (long)kN * kC) return;
  W.agg[idx] = 0.0f;
  int n = (int)(idx / kC), c = (int)(idx % kC);
  if (c < 8) W.chi_nxt[(long)n * 8 + c] = W.chi[(long)n * 8 + c];
}

// Fused q|k|v projection: [N,128] @ [128,384], WMMA tiles.
__global__ __launch_bounds__(128) void k_gemm_qkv(Wsp W, int l) {
  int lane = threadIdx.x & 31, wv = threadIdx.x >> 5;
  int wid = blockIdx.x * 4 + wv;
  int rt = wid / 24, ct = wid % 24;
  const _Float16* A = W.nf16 + (long)rt * 16 * kC;
  v8f acc;
#pragma unroll
  for (int i = 0; i < 8; ++i) acc[i] = 0.0f;
#pragma unroll
  for (int kc = 0; kc < 4; ++kc)
    acc = wmma16(load_a16(A, kC, kc * 32), load_bt16(W.Wqkvt[l], kC, ct, kc * 32), acc);
  int hi = lane >> 4, col = ct * 16 + (lane & 15);
#pragma unroll
  for (int i = 0; i < 8; ++i) {
    long row = (long)rt * 16 + i + 8 * hi;
    W.qkv[row * 384 + col] = acc[i];
  }
}

// Small geometric projections qg/kg: [N,128]@[128,2] each (VALU).
__global__ __launch_bounds__(256) void k_qgkg(Params P, Wsp W, int l) {
  int idx = blockIdx.x * 256 + threadIdx.x;
  if (idx >= kN * 4) return;
  int n = idx >> 2, o = idx & 3, j = o & 1;
  const float* Wm = ((o < 2) ? P.Wqg : P.Wkg) + l * 256;
  const float* x = W.nf + (long)n * kC;
  float s = 0.0f;
  for (int k = 0; k < kC; ++k) s += x[k] * Wm[k * 2 + j];
  ((o < 2) ? W.qg : W.kg)[(long)n * 2 + j] = s;
}

// ---------------------------------------------------------------------------
// Edge block: per wave, a 16-edge tile.
//   fb filter : WMMA 16x32x64 (radial) + VALU sph hidden -> WMMA 16x128x128
//   attention : per-edge q*w*k dot, atomic scatter of alpha*v into agg
//   gb filter : WMMA 16x32x64 + VALU -> 2 outputs -> atomic chi scatter
// ---------------------------------------------------------------------------
__global__ __launch_bounds__(128) void k_edge_block(Params P, Wsp W, int l) {
  __shared__ _Float16 sH[4][16][128];   // silu hidden, f16 (WMMA A staging)
  __shared__ float    sW_[4][16][128];  // attention filter w, f32
  __shared__ float    sCS[4][16][2];
  __shared__ float    sSH[4][16][8];
  __shared__ float    sCut[4][16];
  __shared__ int      sS[4][16], sR[4][16];

  int lane = threadIdx.x & 31, wv = threadIdx.x >> 5;
  long e0 = ((long)blockIdx.x * 4 + wv) * 16;

  // phase 0: gather edge meta + chi_ij per-degree norms
  if (lane < 16) {
    long e = e0 + lane;
    int s = P.senders[e], r = P.receivers[e];
    sS[wv][lane] = s; sR[wv][lane] = r; sCut[wv][lane] = P.cutoffs[e];
    float d1 = 1e-8f, d2 = 1e-8f;
#pragma unroll
    for (int c = 0; c < 8; ++c) {
      float cij = W.chi[(long)s * 8 + c] - W.chi[(long)r * 8 + c];
      if (c < 3) d1 += cij * cij; else d2 += cij * cij;
      sSH[wv][lane][c] = W.sh[e * 8 + c];
    }
    sCS[wv][lane][0] = sqrtf(d1); sCS[wv][lane][1] = sqrtf(d2);
  }
  __syncthreads();

  const int hi = lane >> 4, nlo = lane & 15;
  v16h aR = load_a16(W.ef16 + e0 * kR, kR, 0);

  // phase 1: fb radial hidden (4 WMMAs, K=32)
  const float* fb_b1 = P.fb_rad_b1 + l * kHID;
#pragma unroll
  for (int nt = 0; nt < 4; ++nt) {
    int col = nt * 16 + nlo;
    v8f acc; float b = fb_b1[col];
#pragma unroll
    for (int i = 0; i < 8; ++i) acc[i] = b;
    acc = wmma16(aR, load_bt16(W.fbW1t[l], kR, nt, 0), acc);
#pragma unroll
    for (int i = 0; i < 8; ++i) sH[wv][i + 8 * hi][col] = (_Float16)silu(acc[i]);
  }
  // phase 2: fb spherical hidden (K=2 -> VALU), cols 64..127
  {
    int eix = lane >> 1, j0 = (lane & 1) * 32;
    float c0 = sCS[wv][eix][0], c1 = sCS[wv][eix][1];
    const float* W1 = P.fb_sph_W1 + l * 2 * kHID;
    const float* b1 = P.fb_sph_b1 + l * kHID;
#pragma unroll
    for (int j = 0; j < 32; ++j) {
      int jj = j0 + j;
      sH[wv][eix][64 + jj] = (_Float16)silu(c0 * W1[jj] + c1 * W1[kHID + jj] + b1[jj]);
    }
  }
  __syncthreads();
  // phase 3: w = hidden(16x128) @ W2cat(128x128) + b  (32 WMMAs)
#pragma unroll
  for (int nt = 0; nt < 8; ++nt) {
    int col = nt * 16 + nlo;
    v8f acc; float b = W.fb_b2sum[l][col];
#pragma unroll
    for (int i = 0; i < 8; ++i) acc[i] = b;
#pragma unroll
    for (int kc = 0; kc < 4; ++kc)
      acc = wmma16(load_a16(&sH[wv][0][0], 128, kc * 32),
                   load_bt16(W.fbW2t[l], 128, nt, kc * 32), acc);
#pragma unroll
    for (int i = 0; i < 8; ++i) sW_[wv][i + 8 * hi][col] = acc[i];
  }
  __syncthreads();
  // phase 4: attention score + message scatter (2 heads per lane)
  {
    int eix = lane >> 1, hp = lane & 1;
    int s = sS[wv][eix], r = sR[wv][eix];
    float scale = sCut[wv][eix] * (1.0f / 16.0f) * 0.17677669529f; // /AVG_NEIGH /sqrt(DH)
#pragma unroll
    for (int hh = 0; hh < 2; ++hh) {
      int h = 2 * hp + hh;
      const float* q = W.qkv + (long)r * 384 + h * 32;
      const float* k = W.qkv + (long)s * 384 + 128 + h * 32;
      const float* v = W.qkv + (long)s * 384 + 256 + h * 32;
      float al = 0.0f;
#pragma unroll
      for (int d = 0; d < 32; ++d) al += q[d] * sW_[wv][eix][h * 32 + d] * k[d];
      al *= scale;
      float* dst = W.agg + (long)r * kC + h * 32;
#pragma unroll
      for (int d = 0; d < 32; ++d) atomicAdd(&dst[d], al * v[d]);
    }
  }
  __syncthreads();
  // phase 5: geometric filter (reuse sH)
  const float* gb_b1 = P.gb_rad_b1 + l * kHID;
#pragma unroll
  for (int nt = 0; nt < 4; ++nt) {
    int col = nt * 16 + nlo;
    v8f acc; float b = gb_b1[col];
#pragma unroll
    for (int i = 0; i < 8; ++i) acc[i] = b;
    acc = wmma16(aR, load_bt16(W.gbW1t[l], kR, nt, 0), acc);
#pragma unroll
    for (int i = 0; i < 8; ++i) sH[wv][i + 8 * hi][col] = (_Float16)silu(acc[i]);
  }
  {
    int eix = lane >> 1, j0 = (lane & 1) * 32;
    float c0 = sCS[wv][eix][0], c1 = sCS[wv][eix][1];
    const float* W1 = P.gb_sph_W1 + l * 2 * kHID;
    const float* b1 = P.gb_sph_b1 + l * kHID;
#pragma unroll
    for (int j = 0; j < 32; ++j) {
      int jj = j0 + j;
      sH[wv][eix][64 + jj] = (_Float16)silu(c0 * W1[jj] + c1 * W1[kHID + jj] + b1[jj]);
    }
  }
  __syncthreads();
  {
    int eix = lane >> 1, j = lane & 1;
    float wg = W.gb_b2sum[l][j];
    const float* W2 = W.gbW2cat[l];
#pragma unroll
    for (int kk = 0; kk < 128; ++kk) wg += (float)sH[wv][eix][kk] * W2[kk * 2 + j];
    int s = sS[wv][eix], r = sR[wv][eix];
    float ag = W.qg[(long)r * 2 + j] * wg * W.kg[(long)s * 2 + j] *
               sCut[wv][eix] * (1.0f / 16.0f);
    int cA = (j == 0) ? 0 : 3, cB = (j == 0) ? 3 : 8;
    for (int c = cA; c < cB; ++c)
      atomicAdd(&W.chi_nxt[(long)r * 8 + c], ag * sSH[wv][eix][c]);
  }
}

__global__ __launch_bounds__(256) void k_cvt_agg(Wsp W) {
  long idx = (long)blockIdx.x * 256 + threadIdx.x;
  if (idx >= (long)kN * kC) return;
  W.agg16[idx] = (_Float16)W.agg[idx];
}

// nf += agg @ Wo  (WMMA)
__global__ __launch_bounds__(128) void k_gemm_wo(Wsp W, int l) {
  int lane = threadIdx.x & 31, wv = threadIdx.x >> 5;
  int wid = blockIdx.x * 4 + wv;
  int rt = wid / 8, ct = wid % 8;
  v8f acc;
#pragma unroll
  for (int i = 0; i < 8; ++i) acc[i] = 0.0f;
#pragma unroll
  for (int kc = 0; kc < 4; ++kc)
    acc = wmma16(load_a16(W.agg16 + (long)rt * 16 * kC, kC, kc * 32),
                 load_bt16(W.Wot[l], kC, ct, kc * 32), acc);
  int hi = lane >> 4, col = ct * 16 + (lane & 15);
#pragma unroll
  for (int i = 0; i < 8; ++i) {
    long row = (long)rt * 16 + i + 8 * hi;
    W.nf[row * kC + col] += acc[i];
  }
}

// Interaction block: y = [nf, deg_norm(chi)] @ ib_W + b; update nf, gate chi.
__global__ __launch_bounds__(256) void k_interaction(Params P, Wsp W, int l) {
  __shared__ float xin[8][132];
  int lane = threadIdx.x & 31, wv = threadIdx.x >> 5;
  long n = (long)blockIdx.x * 8 + wv;
  for (int idx = lane; idx < kC; idx += 32) xin[wv][idx] = W.nf[n * kC + idx];
  if (lane == 0) {
    float d1 = 1e-8f, d2 = 1e-8f;
    for (int c = 0; c < 8; ++c) {
      float v = W.chi_nxt[n * 8 + c];
      if (c < 3) d1 += v * v; else d2 += v * v;
    }
    xin[wv][128] = sqrtf(d1); xin[wv][129] = sqrtf(d2);
  }
  __syncthreads();
  const float* ibW = P.ib_W + (long)l * 130 * 130;
  const float* ibb = P.ib_b + l * 130;
  for (int j = lane; j < 130; j += 32) {
    float y = ibb[j];
    for (int k = 0; k < 130; ++k) y += xin[wv][k] * ibW[k * 130 + j];
    if (j < kC) {
      float v = xin[wv][j] + y;
      W.nf[n * kC + j] = v;
      W.nf16[n * kC + j] = (_Float16)v;
    } else if (j == 128) {
      for (int c = 0; c < 3; ++c) W.chi[n * 8 + c] = W.chi_nxt[n * 8 + c] * (1.0f + y);
    } else {
      for (int c = 3; c < 8; ++c) W.chi[n * 8 + c] = W.chi_nxt[n * 8 + c] * (1.0f + y);
    }
  }
}

// Readout: silu(nf @ W1 + b1) @ W2 + b2, one 16-node tile per wave.
__global__ __launch_bounds__(32) void k_readout(Params P, Wsp W, float* out) {
  __shared__ float sHh[16][128];
  int lane = threadIdx.x;
  int rt = blockIdx.x;
  const _Float16* A = W.nf16 + (long)rt * 16 * kC;
  int hi = lane >> 4, nlo = lane & 15;
#pragma unroll
  for (int nt = 0; nt < 8; ++nt) {
    int col = nt * 16 + nlo;
    v8f acc; float b = P.out_b1[col];
#pragma unroll
    for (int i = 0; i < 8; ++i) acc[i] = b;
#pragma unroll
    for (int kc = 0; kc < 4; ++kc)
      acc = wmma16(load_a16(A, kC, kc * 32), load_bt16(W.outW1t, kC, nt, kc * 32), acc);
#pragma unroll
    for (int i = 0; i < 8; ++i) sHh[i + 8 * hi][col] = silu(acc[i]);
  }
  __syncthreads();
  if (lane < 16) {
    float s = P.out_b2[0];
    for (int k = 0; k < kC; ++k) s += sHh[lane][k] * P.out_W2[k];
    out[(long)rt * 16 + lane] = s;
  }
}

// ---------------------------------------------------------------------------
extern "C" void kernel_launch(void* const* d_in, const int* in_sizes, int n_in,
                              void* d_out, int out_size, void* d_ws, size_t ws_size,
                              hipStream_t stream) {
  (void)in_sizes; (void)n_in; (void)out_size; (void)ws_size;
  // d_in order: top-level dict insertion order, then params tree leaves in
  // sorted-key order (jax tree flatten convention).
  Params P;
  P.edge_vectors = (const float*)d_in[0];
  P.distances    = (const float*)d_in[1];
  P.cutoffs      = (const float*)d_in[2];
  P.species      = (const int*)d_in[3];
  P.senders      = (const int*)d_in[4];
  P.receivers    = (const int*)d_in[5];
  P.embed        = (const float*)d_in[6];
  P.Wk  = (const float*)d_in[7];   P.Wkg = (const float*)d_in[8];
  P.Wo  = (const float*)d_in[9];   P.Wq  = (const float*)d_in[10];
  P.Wqg = (const float*)d_in[11];  P.Wv  = (const float*)d_in[12];
  P.fb_rad_W1 = (const float*)d_in[13]; P.fb_rad_W2 = (const float*)d_in[14];
  P.fb_rad_b1 = (const float*)d_in[15]; P.fb_rad_b2 = (const float*)d_in[16];
  P.fb_sph_W1 = (const float*)d_in[17]; P.fb_sph_W2 = (const float*)d_in[18];
  P.fb_sph_b1 = (const float*)d_in[19]; P.fb_sph_b2 = (const float*)d_in[20];
  P.gb_rad_W1 = (const float*)d_in[21]; P.gb_rad_W2 = (const float*)d_in[22];
  P.gb_rad_b1 = (const float*)d_in[23]; P.gb_rad_b2 = (const float*)d_in[24];
  P.gb_sph_W1 = (const float*)d_in[25]; P.gb_sph_W2 = (const float*)d_in[26];
  P.gb_sph_b1 = (const float*)d_in[27]; P.gb_sph_b2 = (const float*)d_in[28];
  P.ib_W  = (const float*)d_in[29]; P.ib_b  = (const float*)d_in[30];
  P.out_W1 = (const float*)d_in[31]; P.out_W2 = (const float*)d_in[32];
  P.out_b1 = (const float*)d_in[33]; P.out_b2 = (const float*)d_in[34];

  char* p = (char*)d_ws;
  auto alloc = [&](size_t bytes) -> char* {
    char* r = p; p += (bytes + 255) & ~(size_t)255; return r;
  };
  Wsp W;
  W.ef16   = (_Float16*)alloc((size_t)kE * kR * 2);
  W.sh     = (float*)alloc((size_t)kE * 8 * 4);
  W.nf     = (float*)alloc((size_t)kN * kC * 4);
  W.nf16   = (_Float16*)alloc((size_t)kN * kC * 2);
  W.qkv    = (float*)alloc((size_t)kN * 384 * 4);
  W.qg     = (float*)alloc((size_t)kN * 2 * 4);
  W.kg     = (float*)alloc((size_t)kN * 2 * 4);
  W.agg    = (float*)alloc((size_t)kN * kC * 4);
  W.agg16  = (_Float16*)alloc((size_t)kN * kC * 2);
  W.chi    = (float*)alloc((size_t)kN * 8 * 4);
  W.chi_nxt= (float*)alloc((size_t)kN * 8 * 4);
  for (int l = 0; l < 2; ++l) {
    W.fbW1t[l]    = (_Float16*)alloc(64 * 32 * 2);
    W.gbW1t[l]    = (_Float16*)alloc(64 * 32 * 2);
    W.fbW2t[l]    = (_Float16*)alloc(128 * 128 * 2);
    W.Wqkvt[l]    = (_Float16*)alloc(384 * 128 * 2);
    W.Wot[l]      = (_Float16*)alloc(128 * 128 * 2);
    W.fb_b2sum[l] = (float*)alloc(128 * 4);
    W.gb_b2sum[l] = (float*)alloc(2 * 4);
    W.gbW2cat[l]  = (float*)alloc(128 * 2 * 4);
  }
  W.outW1t = (_Float16*)alloc(128 * 128 * 2);

  const int NC_BLK = (kN * kC) / 256;                 // 15000
  k_prep_weights<<<(2 * 86402 + 16384 + 255) / 256, 256, 0, stream>>>(P, W);
  k_prep_nodes  <<<NC_BLK, 256, 0, stream>>>(P, W);
  k_prep_edges  <<<kE / 256, 256, 0, stream>>>(P, W); // 1875

  for (int l = 0; l < 2; ++l) {
    k_begin_layer<<<NC_BLK, 256, 0, stream>>>(W);
    k_gemm_qkv   <<<(kN / 16) * 24 / 4, 128, 0, stream>>>(W, l);   // 11250
    k_qgkg       <<<(kN * 4 + 255) / 256, 256, 0, stream>>>(P, W, l);
    k_edge_block <<<kE / 64, 128, 0, stream>>>(P, W, l);           // 7500
    k_cvt_agg    <<<NC_BLK, 256, 0, stream>>>(W);
    k_gemm_wo    <<<(kN / 16) * 8 / 4, 128, 0, stream>>>(W, l);    // 3750
    k_interaction<<<kN / 8, 256, 0, stream>>>(P, W, l);            // 3750
  }
  k_readout<<<kN / 16, 32, 0, stream>>>(P, W, (float*)d_out);      // 1875
}
